// QuantizedEncoderFromSpec_50302656971349
// MI455X (gfx1250) — matrix operand order repaired and checked by reading, not compiled, not run on hardware
//
#include <hip/hip_runtime.h>
#include <hip/hip_bf16.h>
#include <cstdint>

// ---------------------------------------------------------------------------
// QuantizedEncoder for MI455X (gfx1250, wave32, WMMA).
// All activations kept in bf16 so the 134MB `h`/`z` tensors live in the 192MB
// L2; GEMMs use v_wmma_f32_16x16x32_bf16 with f32 accumulate. A-tiles are
// staged with GLOBAL_LOAD_ASYNC_TO_LDS_B128 when available.
// ---------------------------------------------------------------------------

#define B_   32768
#define DL_  512
#define G_   8
#define CG_  256
#define NE_  512
#define GC_  (G_ * CG_)   // 2048
#define OUT_ 3

typedef unsigned short u16;
typedef __attribute__((ext_vector_type(16))) __bf16 v16bf;
typedef __attribute__((ext_vector_type(8)))  float  v8f;
typedef int v4i_t __attribute__((ext_vector_type(4)));

__device__ __forceinline__ u16 f2bf(float f) {
  union { float f; unsigned u; } v; v.f = f;
  unsigned u = v.u;
  u += 0x7fffu + ((u >> 16) & 1u);      // round-to-nearest-even
  return (u16)(u >> 16);
}
__device__ __forceinline__ float bf2f(u16 h) {
  union { unsigned u; float f; } v; v.u = ((unsigned)h) << 16;
  return v.f;
}

// ---------------------------------------------------------------------------
// Generic (optionally grouped) bf16 GEMM:  C_g = A_g * B_g (+bias) (+addF)
//   A : (M, G*K) bf16 row-major, group g occupies cols [g*K, (g+1)*K)
//   Bw: (G, K, N) bf16
//   C : (M, G*N), f32 and/or bf16 outputs; addF optional f32 addend (same idx)
// Tile 64x64x32, 8 waves (wave32), each wave -> 32x16 via two 16x16x32 WMMAs.
// ---------------------------------------------------------------------------
#define BM 64
#define BN 64
#define BK 32

__launch_bounds__(256, 2)
__global__ void k_gemm_bf16(const u16* __restrict__ A, int lda,
                            const u16* __restrict__ Bw,
                            const float* __restrict__ bias,
                            float* __restrict__ Cf, u16* __restrict__ Cb, int ldc,
                            const float* __restrict__ addF,
                            int M, int N, int K)
{
  __shared__ u16 sA[BM * BK];          // [64][32] row-major
  __shared__ u16 sB[(BK / 2) * BN * 2];// [16 pairs][64 cols][2] -> frag dword loads

  const int tid  = threadIdx.x;
  const int g    = blockIdx.z;
  const int m0   = blockIdx.y * BM;
  const int n0   = blockIdx.x * BN;
  const long aOff = (long)g * K;
  const long cOff = (long)g * N;
  const u16* Bg   = Bw + (long)g * K * N;

  const int wid = tid >> 5, lane = tid & 31;
  const int wm = wid & 1, wn = wid >> 1;       // wave rows: wm*32, wave cols: wn*16
  const int lhalf = lane >> 4, l15 = lane & 15;

  v8f acc0 = {}; v8f acc1 = {};

  const int ktiles = (K + BK - 1) / BK;
  const int arow = tid >> 2, ac8 = (tid & 3) * 8;  // A loader: 64 rows x 4 chunks of 8
  const int brow = tid >> 3, bc8 = (tid & 7) * 8;  // B slow loader
  const int bp   = tid >> 4, bn4 = (tid & 15) * 4; // B fast loader: pair-row x 4 cols

  // workgroup-uniform fast-path eligibility (independent of kt except k-range)
  const bool alignA = ((aOff & 7) == 0) && ((lda & 7) == 0) && (m0 + BM <= M);
  const bool alignB = ((N & 3) == 0) && (n0 + BN <= N);

  for (int kt = 0; kt < ktiles; ++kt) {
    const int k0 = kt * BK;
    const bool kFull = (k0 + BK <= K);
    bool usedAsync = false;

    if (alignA && kFull) {             // ---- fast A: 16B per thread ----
      const u16* gp = &A[(long)(m0 + arow) * lda + aOff + k0 + ac8];
#if defined(__gfx1250__) && __has_builtin(__builtin_amdgcn_global_load_async_to_lds_b128)
      __builtin_amdgcn_global_load_async_to_lds_b128(
          (__attribute__((address_space(1))) v4i_t*)gp,
          (__attribute__((address_space(3))) v4i_t*)&sA[arow * BK + ac8],
          0, 0);
      usedAsync = true;
#else
      *(uint4*)&sA[arow * BK + ac8] = *(const uint4*)gp;
#endif
    } else {                           // ---- slow A (edge tiles) ----
      const int m = m0 + arow;
#pragma unroll
      for (int j = 0; j < 8; ++j) {
        const int k = k0 + ac8 + j;
        u16 val = 0;
        if (m < M && k < K) val = A[(long)m * lda + aOff + k];
        sA[arow * BK + ac8 + j] = val;
      }
    }

    if (alignB && kFull) {             // ---- fast B: interleave 2 k-rows x 4 cols ----
      const int k = k0 + 2 * bp;
      const u16* gp = &Bg[(long)k * N + n0 + bn4];
      const uint2 ra = *(const uint2*)gp;        // row k   : 4 bf16
      const uint2 rb = *(const uint2*)(gp + N);  // row k+1 : 4 bf16
      uint4 w;
      w.x = (ra.x & 0xffffu) | (rb.x << 16);
      w.y = (ra.x >> 16) | (rb.x & 0xffff0000u);
      w.z = (ra.y & 0xffffu) | (rb.y << 16);
      w.w = (ra.y >> 16) | (rb.y & 0xffff0000u);
      *(uint4*)&sB[bp * (BN * 2) + bn4 * 2] = w;
    } else {                           // ---- slow B (edge tiles) ----
      const int k = k0 + brow;
#pragma unroll
      for (int j = 0; j < 8; ++j) {
        const int n = n0 + bc8 + j;
        u16 val = 0;
        if (k < K && n < N) val = Bg[(long)k * N + n];
        sB[(brow >> 1) * (BN * 2) + (bc8 + j) * 2 + (brow & 1)] = val;
      }
    }

    if (usedAsync) {
#if __has_builtin(__builtin_amdgcn_s_wait_asynccnt)
      __builtin_amdgcn_s_wait_asynccnt(0);
#else
      asm volatile("s_wait_asynccnt 0x0" ::: "memory");
#endif
    }
    __syncthreads();

    // B fragment: lane = (n = l15, K-half = lhalf*16); VGPR v holds K pair 2v
    union { v16bf v; unsigned d[8]; } bfr;
    {
      const int nl = wn * 16 + l15;
      const int pb = lhalf * 8;
#pragma unroll
      for (int v = 0; v < 8; ++v)
        bfr.d[v] = *(const unsigned*)&sB[(pb + v) * (BN * 2) + nl * 2];
    }
    // A fragments (same k order as B): lane m=l15, K-half = lhalf*16 (contiguous)
    union { v16bf v; unsigned d[8]; } af0, af1;
    {
      const int r0 = wm * 32 + l15;
      const int kb = lhalf * 16;
#pragma unroll
      for (int v = 0; v < 4; ++v) {
        af0.d[v]     = *(const unsigned*)&sA[r0 * BK + kb + 2 * v];
        af0.d[v + 4] = *(const unsigned*)&sA[r0 * BK + 8 + kb + 2 * v];
      }
      const int r1 = r0 + 16;
#pragma unroll
      for (int v = 0; v < 4; ++v) {
        af1.d[v]     = *(const unsigned*)&sA[r1 * BK + kb + 2 * v];
        af1.d[v + 4] = *(const unsigned*)&sA[r1 * BK + 8 + kb + 2 * v];
      }
    }
    acc0 = __builtin_amdgcn_wmma_f32_16x16x32_bf16(false, af0.v, false, bfr.v,
                                                   (short)0, acc0, false, false);
    acc1 = __builtin_amdgcn_wmma_f32_16x16x32_bf16(false, af1.v, false, bfr.v,
                                                   (short)0, acc1, false, false);
    __syncthreads();
  }

  // Epilogue (C/D layout: VGPR r -> M = r (+8 for lane-half 1), N = lane&15)
  union { v8f v; float f[8]; } o0, o1; o0.v = acc0; o1.v = acc1;
  const int ng = n0 + wn * 16 + l15;
#pragma unroll
  for (int fm = 0; fm < 2; ++fm) {
#pragma unroll
    for (int r = 0; r < 8; ++r) {
      const int mg = m0 + wm * 32 + fm * 16 + lhalf * 8 + r;
      if (mg < M && ng < N) {
        float val = fm ? o1.f[r] : o0.f[r];
        if (bias) val += bias[(long)g * N + ng];
        const long cidx = (long)mg * ldc + cOff + ng;
        if (addF) val += addF[cidx];
        if (Cf) Cf[cidx] = val;
        if (Cb) Cb[cidx] = f2bf(val);
      }
    }
  }
}

// ---------------------------------------------------------------------------
// BatchNorm stats: per-channel sum & sumsq over M rows (atomic partials).
// sums[0..C) = sum, sums[C..2C) = sumsq.  Input f32 OR bf16.
// ---------------------------------------------------------------------------
__global__ void k_stats(const float* __restrict__ Xf, const u16* __restrict__ Xb,
                        int M, int C, float* __restrict__ sums)
{
  const int c = blockIdx.x * blockDim.x + threadIdx.x;
  if (c >= C) return;
  const int chunk = (M + gridDim.y - 1) / gridDim.y;
  const int mA = blockIdx.y * chunk;
  const int mB = min(M, mA + chunk);
  float s = 0.f, s2 = 0.f;
  for (int m = mA; m < mB; ++m) {
    const float v = Xf ? Xf[(long)m * C + c] : bf2f(Xb[(long)m * C + c]);
    s += v; s2 += v * v;
  }
  atomicAdd(&sums[c], s);
  atomicAdd(&sums[C + c], s2);
}

// ---------------------------------------------------------------------------
// y = act(bn(x)) -> bf16.  act: 0 none, 1 relu, 2 lrelu(0.2), 3 lrelu∘lrelu.
// sums==null skips BN (pure activation).
// ---------------------------------------------------------------------------
__global__ void k_bnact(const float* __restrict__ Xf, const u16* __restrict__ Xb,
                        int M, int C, const float* __restrict__ sums,
                        const float* __restrict__ gamma, const float* __restrict__ beta,
                        int act, u16* __restrict__ Yb)
{
  const long total = (long)M * C;
  const long stride = (long)gridDim.x * blockDim.x;
  const float invM = 1.0f / (float)M;
  for (long i = blockIdx.x * (long)blockDim.x + threadIdx.x; i < total; i += stride) {
    const int c = (int)(i % C);
    float v = Xf ? Xf[i] : bf2f(Xb[i]);
    if (sums) {
      const float mean = sums[c] * invM;
      const float var  = sums[C + c] * invM - mean * mean;
      v = (v - mean) * __frsqrt_rn(var + 1e-5f);
      v = v * gamma[c] + beta[c];
    }
    if      (act == 1) v = v > 0.f ? v : 0.f;
    else if (act == 2) v = v >= 0.f ? v : 0.2f * v;
    else if (act == 3) v = v >= 0.f ? v : 0.04f * v;
    Yb[i] = f2bf(v);
  }
}

// (G,R,C) f32 -> (G,C,R) bf16
__global__ void k_transpose_bf16(const float* __restrict__ in, u16* __restrict__ out,
                                 int G, int R, int C)
{
  const long total = (long)G * R * C;
  const long stride = (long)gridDim.x * blockDim.x;
  for (long i = blockIdx.x * (long)blockDim.x + threadIdx.x; i < total; i += stride) {
    const long g = i / ((long)R * C);
    const long rem = i - g * (long)R * C;
    const int r = (int)(rem / C), c = (int)(rem % C);
    out[g * (long)R * C + (long)c * R + r] = f2bf(in[i]);
  }
}

__global__ void k_f32_to_bf16(const float* __restrict__ in, u16* __restrict__ out, long n)
{
  const long stride = (long)gridDim.x * blockDim.x;
  for (long i = blockIdx.x * (long)blockDim.x + threadIdx.x; i < n; i += stride)
    out[i] = f2bf(in[i]);
}

// e2[n] = sum_c embed[c,n]^2   (embed: 2048 x 512 f32)
__global__ void k_e2(const float* __restrict__ embed, float* __restrict__ e2)
{
  const int n = blockIdx.x * blockDim.x + threadIdx.x;
  if (n >= NE_) return;
  float s = 0.f;
  for (int c = 0; c < GC_; ++c) { const float v = embed[(long)c * NE_ + n]; s += v * v; }
  e2[n] = s;
}

// ---------------------------------------------------------------------------
// One wave32 per row: argmin_n(e2[n] - 2*score[n]), gather q=embed[:,ind] to
// bf16 z0, accumulate diff = mean((q - h)^2), emit ind (as float).
// ---------------------------------------------------------------------------
__launch_bounds__(256)
__global__ void k_argmin(const float* __restrict__ scores, const float* __restrict__ e2,
                         const float* __restrict__ embed, const u16* __restrict__ hb,
                         u16* __restrict__ z0, float* __restrict__ diffAcc,
                         float* __restrict__ indOut, int Bn)
{
  const int gt = blockIdx.x * blockDim.x + threadIdx.x;
  const int row = gt >> 5, lane = gt & 31;
  if (row >= Bn) return;
  const float* sr = scores + (long)row * NE_;
  float best = 3.4e38f; int bi = 0x7fffffff;
  for (int n = lane; n < NE_; n += 32) {
    const float d = e2[n] - 2.0f * sr[n];
    if (d < best) { best = d; bi = n; }
  }
  for (int off = 16; off; off >>= 1) {
    const float ob = __shfl_xor(best, off, 32);
    const int   oi = __shfl_xor(bi, off, 32);
    if (ob < best || (ob == best && oi < bi)) { best = ob; bi = oi; }
  }
  const u16* hr = hb + (long)row * GC_;
  u16* zr = z0 + (long)row * GC_;
  float acc = 0.f;
  for (int c = lane; c < GC_; c += 32) {
    const float q = embed[(long)c * NE_ + bi];
    zr[c] = f2bf(q);
    const float dh = q - bf2f(hr[c]);
    acc += dh * dh;
  }
  for (int off = 16; off; off >>= 1) acc += __shfl_xor(acc, off, 32);
  if (lane == 0) {
    atomicAdd(diffAcc, acc * (1.0f / ((float)Bn * (float)GC_)));
    indOut[row] = (float)bi;
  }
}

// ---------------------------------------------------------------------------
extern "C" void kernel_launch(void* const* d_in, const int* in_sizes, int n_in,
                              void* d_out, int out_size, void* d_ws, size_t ws_size,
                              hipStream_t stream)
{
  (void)in_sizes; (void)n_in; (void)out_size; (void)ws_size;
  const float* x    = (const float*)d_in[0];
  const float* g0   = (const float*)d_in[1];
  const float* b0   = (const float*)d_in[2];
  const float* Wc   = (const float*)d_in[3];
  const float* bc   = (const float*)d_in[4];
  const float* emb  = (const float*)d_in[5];
  const float* Wout = (const float*)d_in[60];
  const float* bout = (const float*)d_in[61];
  static const int CI[6] = {256, 128, 64, 32, 16, 8};
  static const int CO[6] = {128, 64, 32, 16, 8, 4};

  char* ws = (char*)d_ws;
  size_t off = 0;
  auto alloc = [&](size_t bytes) -> size_t {
    size_t o = off; off = (off + bytes + 255) & ~(size_t)255; return o;
  };
  // small buffers
  const size_t oStats = alloc(14 * 4096 * sizeof(float));  // 14 BN-stat slots
  const size_t oE2    = alloc(NE_ * sizeof(float));
  const size_t oWCt   = alloc((size_t)DL_ * GC_ * 2);
  const size_t oEMB   = alloc((size_t)GC_ * NE_ * 2);
  const size_t oWOt   = alloc((size_t)G_ * 4 * OUT_ * 2);
  size_t oW0t[6], oW1t[6], oWst[6];
  for (int p = 0; p < 6; ++p) {
    oW0t[p] = alloc((size_t)G_ * CI[p] * CO[p] * 2);
    oW1t[p] = alloc((size_t)G_ * CO[p] * CO[p] * 2);
    oWst[p] = alloc((size_t)G_ * CI[p] * CO[p] * 2);
  }
  // big bf16 activation buffers (L2-friendly): ~582 MB total workspace
  const size_t oH = alloc((size_t)B_ * GC_ * 2);        // h / z ping
  const size_t oZ = alloc((size_t)B_ * GC_ * 2);        // q / z pong
  const size_t oA = alloc((size_t)B_ * GC_ * 2);        // activation scratch
  const size_t oN = alloc((size_t)B_ * (GC_ / 2) * 2);  // net
  const size_t oT = alloc((size_t)B_ * (GC_ / 2) * 4);  // f32 scores / dz tmp

  float* retOut  = (float*)d_out;
  float* diffOut = retOut + (size_t)B_ * G_ * OUT_;
  float* indOut  = diffOut + 1;
  float* stats0  = (float*)(ws + oStats);

  (void)hipMemsetAsync(ws + oStats, 0, 14 * 4096 * sizeof(float), stream);
  (void)hipMemsetAsync(diffOut, 0, sizeof(float), stream);

  // ---- weight prep (f32 -> bf16, transposed to K x N) ----
  k_transpose_bf16<<<256, 256, 0, stream>>>(Wc, (u16*)(ws + oWCt), 1, GC_, DL_);
  k_f32_to_bf16<<<256, 256, 0, stream>>>(emb, (u16*)(ws + oEMB), (long)GC_ * NE_);
  k_transpose_bf16<<<1, 256, 0, stream>>>(Wout, (u16*)(ws + oWOt), G_, OUT_, 4);
  for (int p = 0; p < 6; ++p) {
    const float* W0p = (const float*)d_in[6 + 9 * p + 4];
    const float* W1p = (const float*)d_in[6 + 9 * p + 6];
    const float* Wsp = (const float*)d_in[6 + 9 * p + 8];
    k_transpose_bf16<<<128, 256, 0, stream>>>(W0p, (u16*)(ws + oW0t[p]), G_, CO[p], CI[p]);
    k_transpose_bf16<<<128, 256, 0, stream>>>(W1p, (u16*)(ws + oW1t[p]), G_, CO[p], CO[p]);
    k_transpose_bf16<<<128, 256, 0, stream>>>(Wsp, (u16*)(ws + oWst[p]), G_, CO[p], CI[p]);
  }

  // ---- stage 1: lrelu(bn(x)) -> A ; h = A @ Wc^T + bc -> H (bf16) ----
  k_stats<<<dim3((DL_ + 255) / 256, 128), 256, 0, stream>>>(x, nullptr, B_, DL_, stats0);
  k_bnact<<<4096, 256, 0, stream>>>(x, nullptr, B_, DL_, stats0, g0, b0, 2, (u16*)(ws + oA));
  k_gemm_bf16<<<dim3(GC_ / BN, B_ / BM, 1), 256, 0, stream>>>(
      (u16*)(ws + oA), DL_, (u16*)(ws + oWCt), bc,
      nullptr, (u16*)(ws + oH), GC_, nullptr, B_, GC_, DL_);

  // ---- quantize: scores = h @ embed ; argmin + gather + diff -> Z ----
  k_gemm_bf16<<<dim3(NE_ / BN, B_ / BM, 1), 256, 0, stream>>>(
      (u16*)(ws + oH), GC_, (u16*)(ws + oEMB), nullptr,
      (float*)(ws + oT), nullptr, NE_, nullptr, B_, NE_, GC_);
  k_e2<<<2, 256, 0, stream>>>(emb, (float*)(ws + oE2));
  k_argmin<<<(B_ * 32) / 256, 256, 0, stream>>>(
      (float*)(ws + oT), (float*)(ws + oE2), emb, (u16*)(ws + oH),
      (u16*)(ws + oZ), diffOut, indOut, B_);

  // ---- grouped resnet blocks ----
  u16* zin  = (u16*)(ws + oZ);
  u16* zout = (u16*)(ws + oH);   // h dead after argmin -> reuse as ping buffer
  for (int p = 0; p < 6; ++p) {
    const int ci = CI[p], co = CO[p];
    const int curC = G_ * ci, outC = G_ * co;
    const float* bg0 = (const float*)d_in[6 + 9 * p + 0];
    const float* bb0 = (const float*)d_in[6 + 9 * p + 1];
    const float* bg1 = (const float*)d_in[6 + 9 * p + 2];
    const float* bb1 = (const float*)d_in[6 + 9 * p + 3];
    const float* c0  = (const float*)d_in[6 + 9 * p + 5];
    const float* c1  = (const float*)d_in[6 + 9 * p + 7];
    float* s0 = stats0 + (1 + 2 * p) * 4096;
    float* s1 = stats0 + (2 + 2 * p) * 4096;

    k_stats<<<dim3((curC + 255) / 256, 128), 256, 0, stream>>>(nullptr, zin, B_, curC, s0);
    k_bnact<<<4096, 256, 0, stream>>>(nullptr, zin, B_, curC, s0, bg0, bb0, 1, (u16*)(ws + oA));
    k_gemm_bf16<<<dim3((co + BN - 1) / BN, B_ / BM, G_), 256, 0, stream>>>(
        (u16*)(ws + oA), curC, (u16*)(ws + oW0t[p]), c0,
        nullptr, (u16*)(ws + oN), outC, nullptr, B_, co, ci);

    k_stats<<<dim3((outC + 255) / 256, 128), 256, 0, stream>>>(nullptr, (u16*)(ws + oN), B_, outC, s1);
    k_bnact<<<4096, 256, 0, stream>>>(nullptr, (u16*)(ws + oN), B_, outC, s1, bg1, bb1, 1, (u16*)(ws + oA));
    k_gemm_bf16<<<dim3((co + BN - 1) / BN, B_ / BM, G_), 256, 0, stream>>>(
        (u16*)(ws + oA), outC, (u16*)(ws + oW1t[p]), c1,
        (float*)(ws + oT), nullptr, outC, nullptr, B_, co, co);

    // z_next = z @ Ws^T + dz (f32 addend), stored bf16
    k_gemm_bf16<<<dim3((co + BN - 1) / BN, B_ / BM, G_), 256, 0, stream>>>(
        zin, curC, (u16*)(ws + oWst[p]), nullptr,
        nullptr, zout, outC, (float*)(ws + oT), B_, co, ci);
    u16* t = zin; zin = zout; zout = t;
  }

  // ---- final double lrelu + output conv -> d_out ----
  k_bnact<<<512, 256, 0, stream>>>(nullptr, zin, B_, G_ * CO[5], nullptr, nullptr, nullptr,
                                   3, (u16*)(ws + oA));
  k_gemm_bf16<<<dim3(1, B_ / BM, G_), 256, 0, stream>>>(
      (u16*)(ws + oA), G_ * CO[5], (u16*)(ws + oWOt), bout,
      retOut, nullptr, G_ * OUT_, nullptr, B_, OUT_, 4);
}
